// LSCCM_53910429499672
// MI455X (gfx1250) — compile-verified, matrix operand order checked by reading, not run on hardware
//
#include <hip/hip_runtime.h>

// ---------------------------------------------------------------------------
// CDNA5 (gfx1250) WMMA f16 building blocks (wave32)
// ---------------------------------------------------------------------------
typedef __attribute__((ext_vector_type(16))) _Float16 v16h;
typedef __attribute__((ext_vector_type(8)))  _Float16 v8h;
typedef __attribute__((ext_vector_type(8)))  float    v8f;

// Load a 16x32 f16 fragment in the documented A/B striping:
//   lane l: row = l&15, k-halves at kbase + (l>=16?8:0) and +16 more.
// Both runs are 8 contiguous halfs (16 bytes) -> b128 loads.
__device__ __forceinline__ v16h load_frag(const _Float16* base, int ld, int kbase) {
    const int lane = threadIdx.x & 31;
    const int row  = lane & 15;
    const int kh   = (lane >> 4) * 8;
    const _Float16* p = base + (size_t)row * ld + kbase + kh;
    v8h lo = *reinterpret_cast<const v8h*>(p);
    v8h hi = *reinterpret_cast<const v8h*>(p + 16);
    v16h f;
#pragma unroll
    for (int i = 0; i < 8; ++i) { f[i] = lo[i]; f[i + 8] = hi[i]; }
    return f;
}

__device__ __forceinline__ v8f wmma16(v16h a, v16h b, v8f c) {
    return __builtin_amdgcn_wmma_f32_16x16x32_f16(false, a, false, b,
                                                  (short)0, c, false, false);
}

// ---------------------------------------------------------------------------
// K0: fold BN scales into f16 weights, compute BN shift vectors
// ---------------------------------------------------------------------------
__global__ void k_prep(const float* w_red, const float* sr, const float* br,
                       const float* mr, const float* vr,
                       const float* w_q, const float* w_k, const float* w_v,
                       const float* w_cp,
                       const float* w_res, const float* ss, const float* bs,
                       const float* ms, const float* vs,
                       _Float16* wredh, float* beta_red,
                       _Float16* wqh, _Float16* wkh, _Float16* wvh, _Float16* wcph,
                       _Float16* wresh, float* beta_res) {
    int i = blockIdx.x * blockDim.x + threadIdx.x;
    if (i < 64 * 256) {
        int o = i >> 8;
        float inv = sr[o] * rsqrtf(vr[o] + 1e-5f);
        wredh[i] = (_Float16)(w_red[i] * inv);
    }
    if (i < 64) {
        float inv = sr[i] * rsqrtf(vr[i] + 1e-5f);
        beta_red[i] = br[i] - mr[i] * inv;
    }
    if (i < 64 * 64) {
        wqh[i]  = (_Float16)w_q[i];
        wkh[i]  = (_Float16)w_k[i];
        wvh[i]  = (_Float16)w_v[i];
        wcph[i] = (_Float16)w_cp[i];
    }
    if (i < 256 * 128) {
        int o = i >> 7;
        float inv = ss[o] * rsqrtf(vs[o] + 1e-5f);
        wresh[i] = (_Float16)(w_res[i] * inv);
    }
    if (i < 256) {
        float inv = ss[i] * rsqrtf(vs[i] + 1e-5f);
        beta_res[i] = bs[i] - ms[i] * inv;
    }
}

// ---------------------------------------------------------------------------
// K1: x [b][256][4096] f32  ->  xt [b][4096][256] f16 (LDS-tiled transpose)
// ---------------------------------------------------------------------------
__global__ void k_transpose(const float* __restrict__ x, _Float16* __restrict__ xt) {
    __shared__ float tile[32][33];
    const int b  = blockIdx.z;
    const int n0 = blockIdx.x * 32;
    const int c0 = blockIdx.y * 32;
    const int tx = threadIdx.x, ty = threadIdx.y;   // 32x8 block
    const float* xb = x + (size_t)b * 256 * 4096;
#pragma unroll
    for (int i = 0; i < 4; ++i)
        tile[ty + i * 8][tx] = xb[(size_t)(c0 + ty + i * 8) * 4096 + n0 + tx];
    __syncthreads();
    _Float16* xtb = xt + (size_t)b * 4096 * 256;
#pragma unroll
    for (int i = 0; i < 4; ++i)
        xtb[(size_t)(n0 + ty + i * 8) * 256 + c0 + tx] = (_Float16)tile[tx][ty + i * 8];
}

// ---------------------------------------------------------------------------
// K2: reduce conv 256->64 + BN + SiLU.  xrt[b][n][64] = silu(xt @ wredh^T + beta)
// One wave: 16 spatial rows x all 64 output channels, K=256 in 8 WMMA chunks.
// ---------------------------------------------------------------------------
__global__ void __launch_bounds__(256) k_reduce(const _Float16* __restrict__ xt,
                                                const _Float16* __restrict__ wredh,
                                                const float* __restrict__ beta,
                                                _Float16* __restrict__ xrt) {
    const int b    = blockIdx.y;
    const int wave = threadIdx.x >> 5;
    const int lane = threadIdx.x & 31;
    const int n0   = (blockIdx.x * 8 + wave) * 16;
    const _Float16* A = xt + (size_t)b * 4096 * 256 + (size_t)n0 * 256;
    v8f acc[4] = {};
    for (int kk = 0; kk < 256; kk += 32) {
        __builtin_prefetch(A + ((lane & 15) * 256) + kk + 32, 0, 1);
        v16h a = load_frag(A, 256, kk);
#pragma unroll
        for (int t = 0; t < 4; ++t)
            acc[t] = wmma16(a, load_frag(wredh + t * 16 * 256, 256, kk), acc[t]);
    }
    const int col = lane & 15, hl = lane >> 4;
    _Float16* out = xrt + (size_t)b * 4096 * 64;
#pragma unroll
    for (int t = 0; t < 4; ++t) {
        const int o = t * 16 + col;
        const float bb = beta[o];
#pragma unroll
        for (int r = 0; r < 8; ++r) {
            const int m = n0 + r + hl * 8;
            float v = acc[t][r] + bb;
            float s = v / (1.f + __expf(-v));          // SiLU
            out[(size_t)m * 64 + o] = (_Float16)s;
        }
    }
}

// ---------------------------------------------------------------------------
// K3: 4x4 average pool: xrt[b][4096][64] -> pooled[b][256][64]
// ---------------------------------------------------------------------------
__global__ void k_pool(const _Float16* __restrict__ xrt, _Float16* __restrict__ pooled) {
    int idx = blockIdx.x * blockDim.x + threadIdx.x;
    if (idx >= 16 * 256 * 64) return;
    const int o = idx & 63, p = (idx >> 6) & 255, b = idx >> 14;
    const int py = p >> 4, px = p & 15;
    const _Float16* base = xrt + (size_t)b * 4096 * 64;
    float s = 0.f;
#pragma unroll
    for (int dy = 0; dy < 4; ++dy)
#pragma unroll
        for (int dx = 0; dx < 4; ++dx) {
            const int n = (py * 4 + dy) * 64 + px * 4 + dx;
            s += (float)base[(size_t)n * 64 + o];
        }
    pooled[idx] = (_Float16)(s * (1.f / 16.f));
}

// ---------------------------------------------------------------------------
// K4: generic [rows x 64] @ [64 x 64]^T GEMM, f16 in/out, flexible out layout
// ---------------------------------------------------------------------------
__global__ void __launch_bounds__(256) k_gemm64(const _Float16* __restrict__ A, long a_bs,
                                                const _Float16* __restrict__ W, long w_bs,
                                                _Float16* __restrict__ out, long o_bs,
                                                int o_rs, int o_cs, int rows, float scale) {
    const int b    = blockIdx.y;
    const int wave = threadIdx.x >> 5;
    const int lane = threadIdx.x & 31;
    const int n0   = (blockIdx.x * 8 + wave) * 16;
    if (n0 >= rows) return;
    const _Float16* Ab = A + (size_t)b * a_bs + (size_t)n0 * 64;
    const _Float16* Wb = W + (size_t)b * w_bs;
    v8f acc[4] = {};
#pragma unroll
    for (int kk = 0; kk < 64; kk += 32) {
        v16h a = load_frag(Ab, 64, kk);
#pragma unroll
        for (int t = 0; t < 4; ++t)
            acc[t] = wmma16(a, load_frag(Wb + t * 16 * 64, 64, kk), acc[t]);
    }
    const int col = lane & 15, hl = lane >> 4;
    _Float16* ob = out + (size_t)b * o_bs;
#pragma unroll
    for (int t = 0; t < 4; ++t) {
        const int o = t * 16 + col;
#pragma unroll
        for (int r = 0; r < 8; ++r) {
            const int m = n0 + r + hl * 8;
            ob[(size_t)m * o_rs + (size_t)o * o_cs] = (_Float16)(acc[t][r] * scale);
        }
    }
}

// ---------------------------------------------------------------------------
// K5: spatial attention. Per wave: 16 queries vs all 256 pooled keys.
// scores (16 wmma tiles) -> row softmax via shfl_xor over 16-lane groups ->
// probs staged to LDS (D-layout -> A-layout) -> attn @ V (V stored [c][m]).
// Result goes to fused[b][n][0..63].
// ---------------------------------------------------------------------------
__global__ void __launch_bounds__(256) k_spatial(const _Float16* __restrict__ qt,
                                                 const _Float16* __restrict__ kt,
                                                 const _Float16* __restrict__ vcm,
                                                 _Float16* __restrict__ fused) {
    __shared__ _Float16 attn_lds[8][16][128];
    const int b    = blockIdx.y;
    const int wave = threadIdx.x >> 5;
    const int lane = threadIdx.x & 31;
    const int n0   = (blockIdx.x * 8 + wave) * 16;
    const _Float16* Q = qt  + (size_t)b * 4096 * 64 + (size_t)n0 * 64;
    const _Float16* K = kt  + (size_t)b * 256 * 64;
    const _Float16* V = vcm + (size_t)b * 64 * 256;

    v8f sc[16] = {};
#pragma unroll
    for (int kk = 0; kk < 64; kk += 32) {
        v16h a = load_frag(Q, 64, kk);
#pragma unroll
        for (int t = 0; t < 16; ++t)
            sc[t] = wmma16(a, load_frag(K + t * 16 * 64, 64, kk), sc[t]);
    }
    // row-wise softmax over 256 keys; a row lives across lanes 0-15 (or 16-31)
    float inv_sum[8];
#pragma unroll
    for (int r = 0; r < 8; ++r) {
        float m = -1e30f;
#pragma unroll
        for (int t = 0; t < 16; ++t) m = fmaxf(m, sc[t][r]);
#pragma unroll
        for (int d = 1; d < 16; d <<= 1) m = fmaxf(m, __shfl_xor(m, d, 32));
        float s = 0.f;
#pragma unroll
        for (int t = 0; t < 16; ++t) { float e = __expf(sc[t][r] - m); sc[t][r] = e; s += e; }
#pragma unroll
        for (int d = 1; d < 16; d <<= 1) s += __shfl_xor(s, d, 32);
        inv_sum[r] = 1.f / s;
    }
    const int col = lane & 15, hl = lane >> 4;
    v8f oacc[4] = {};
    for (int phase = 0; phase < 2; ++phase) {
        // stage 8 key-tiles of probabilities into LDS as A-layout [16][128]
#pragma unroll
        for (int t = 0; t < 8; ++t) {
            const int tt = phase * 8 + t;
#pragma unroll
            for (int r = 0; r < 8; ++r)
                attn_lds[wave][r + hl * 8][t * 16 + col] =
                    (_Float16)(sc[tt][r] * inv_sum[r]);
        }
        __syncthreads();
        const _Float16* Al = &attn_lds[wave][0][0];
#pragma unroll
        for (int kk = 0; kk < 128; kk += 32) {
            v16h a = load_frag(Al, 128, kk);
#pragma unroll
            for (int t = 0; t < 4; ++t)
                oacc[t] = wmma16(a, load_frag(V + t * 16 * 256, 256, phase * 128 + kk),
                                 oacc[t]);
        }
        __syncthreads();
    }
    _Float16* fb = fused + (size_t)b * 4096 * 128;
#pragma unroll
    for (int t = 0; t < 4; ++t) {
        const int c = t * 16 + col;
#pragma unroll
        for (int r = 0; r < 8; ++r) {
            const int m = n0 + r + hl * 8;
            fb[(size_t)m * 128 + c] = (_Float16)oacc[t][r];
        }
    }
}

// ---------------------------------------------------------------------------
// K6: channel affinity logits: aff[b][64][64] = feat @ feat^T, K = 4096
// feat stored channel-major in cpcm[b][64][4096]; one wave per 16x16 tile.
// ---------------------------------------------------------------------------
__global__ void __launch_bounds__(256) k_aff(const _Float16* __restrict__ cpcm,
                                             float* __restrict__ aff) {
    const int wid  = blockIdx.x * 8 + (threadIdx.x >> 5);   // 0..255
    const int lane = threadIdx.x & 31;
    const int b = wid >> 4, tile = wid & 15;
    const int mt = tile >> 2, nt = tile & 3;
    const _Float16* C = cpcm + (size_t)b * 64 * 4096;
    const _Float16* Am = C + (size_t)mt * 16 * 4096;
    const _Float16* Bn = C + (size_t)nt * 16 * 4096;
    v8f acc = {};
    for (int kk = 0; kk < 4096; kk += 32) {
        __builtin_prefetch(Am + (lane & 15) * 4096 + kk + 64, 0, 1);
        acc = wmma16(load_frag(Am, 4096, kk), load_frag(Bn, 4096, kk), acc);
    }
    const int col = lane & 15, hl = lane >> 4;
    float* ab = aff + (size_t)b * 64 * 64;
#pragma unroll
    for (int r = 0; r < 8; ++r)
        ab[(size_t)(mt * 16 + r + hl * 8) * 64 + nt * 16 + col] = acc[r];
}

// ---------------------------------------------------------------------------
// K7: softmax over affinity rows (scale 1/sqrt(4096) = 1/64), emit f16
// ---------------------------------------------------------------------------
__global__ void k_affsm(const float* __restrict__ aff, _Float16* __restrict__ afth) {
    const int row = blockIdx.x * blockDim.x + threadIdx.x;    // 16*64 rows
    if (row >= 1024) return;
    const float* a = aff + (size_t)row * 64;
    float m = -1e30f;
    for (int i = 0; i < 64; ++i) m = fmaxf(m, a[i] * (1.f / 64.f));
    float s = 0.f;
    for (int i = 0; i < 64; ++i) s += __expf(a[i] * (1.f / 64.f) - m);
    const float inv = 1.f / s;
    _Float16* o = afth + (size_t)row * 64;
    for (int i = 0; i < 64; ++i)
        o[i] = (_Float16)(__expf(a[i] * (1.f / 64.f) - m) * inv);
}

// ---------------------------------------------------------------------------
// K8: restore conv 128->256 + BN, residual: out = x + gamma*(fused@wres^T + beta)
// One wave: 16 spatial rows x 128 of 256 output channels (2 groups), K=128.
// ---------------------------------------------------------------------------
__global__ void __launch_bounds__(256) k_restore(const _Float16* __restrict__ fused,
                                                 const _Float16* __restrict__ wresh,
                                                 const float* __restrict__ beta,
                                                 const float* __restrict__ x,
                                                 const float* __restrict__ gamma,
                                                 float* __restrict__ out) {
    const int b    = blockIdx.y;
    const int wid  = blockIdx.x * 8 + (threadIdx.x >> 5);    // 0..511
    const int lane = threadIdx.x & 31;
    const int n0 = (wid >> 1) * 16, g = wid & 1;
    const _Float16* A = fused + (size_t)b * 4096 * 128 + (size_t)n0 * 128;
    v8f acc[8] = {};
#pragma unroll
    for (int kk = 0; kk < 128; kk += 32) {
        v16h a = load_frag(A, 128, kk);
#pragma unroll
        for (int t = 0; t < 8; ++t)
            acc[t] = wmma16(a, load_frag(wresh + (size_t)(g * 8 + t) * 16 * 128, 128, kk),
                            acc[t]);
    }
    const float gm = gamma[0];
    const int col = lane & 15, hl = lane >> 4;
    const float* xb = x + (size_t)b * 256 * 4096;
    float*       ob = out + (size_t)b * 256 * 4096;
#pragma unroll
    for (int t = 0; t < 8; ++t) {
        const int co = (g * 8 + t) * 16 + col;
        const float bb = beta[co];
#pragma unroll
        for (int r = 0; r < 8; ++r) {
            const int n = n0 + r + hl * 8;
            const size_t idx = (size_t)co * 4096 + n;
            ob[idx] = xb[idx] + gm * (acc[t][r] + bb);
        }
    }
}

// ---------------------------------------------------------------------------
// Host-side orchestration
// ---------------------------------------------------------------------------
extern "C" void kernel_launch(void* const* d_in, const int* in_sizes, int n_in,
                              void* d_out, int out_size, void* d_ws, size_t ws_size,
                              hipStream_t stream) {
    (void)in_sizes; (void)n_in; (void)out_size; (void)ws_size;
    const float* x      = (const float*)d_in[0];
    const float* w_red  = (const float*)d_in[1];
    const float* sr     = (const float*)d_in[2];
    const float* br     = (const float*)d_in[3];
    const float* mr     = (const float*)d_in[4];
    const float* vr     = (const float*)d_in[5];
    const float* w_q    = (const float*)d_in[6];
    const float* w_k    = (const float*)d_in[7];
    const float* w_v    = (const float*)d_in[8];
    const float* w_cp   = (const float*)d_in[9];
    const float* w_res  = (const float*)d_in[10];
    const float* ss     = (const float*)d_in[11];
    const float* bs     = (const float*)d_in[12];
    const float* ms     = (const float*)d_in[13];
    const float* vs     = (const float*)d_in[14];
    const float* gamma  = (const float*)d_in[15];
    float* out = (float*)d_out;

    // workspace carve-out (256B aligned)
    char* ws = (char*)d_ws;
    auto ah = [&](size_t n) { _Float16* p = (_Float16*)ws; ws += ((n * 2 + 255) / 256) * 256; return p; };
    auto af = [&](size_t n) { float*    p = (float*)ws;    ws += ((n * 4 + 255) / 256) * 256; return p; };

    _Float16* wredh   = ah(64 * 256);
    _Float16* wqh     = ah(64 * 64);
    _Float16* wkh     = ah(64 * 64);
    _Float16* wvh     = ah(64 * 64);
    _Float16* wcph    = ah(64 * 64);
    _Float16* wresh   = ah(256 * 128);
    float*    beta_r  = af(64);
    float*    beta_s  = af(256);
    _Float16* xt      = ah((size_t)16 * 4096 * 256);
    _Float16* xrt     = ah((size_t)16 * 4096 * 64);
    _Float16* pooled  = ah((size_t)16 * 256 * 64);
    _Float16* qt      = ah((size_t)16 * 4096 * 64);
    _Float16* cpt     = ah((size_t)16 * 4096 * 64);   // [n][c]
    _Float16* cpcm    = ah((size_t)16 * 64 * 4096);   // [c][n]
    _Float16* ktile   = ah((size_t)16 * 256 * 64);    // [m][c]
    _Float16* vcm     = ah((size_t)16 * 64 * 256);    // [c][m]
    float*    aff     = af((size_t)16 * 64 * 64);
    _Float16* afth    = ah((size_t)16 * 64 * 64);
    _Float16* fused   = ah((size_t)16 * 4096 * 128);  // [n][spatial|channel]

    // K0: weight prep
    k_prep<<<128, 256, 0, stream>>>(w_red, sr, br, mr, vr, w_q, w_k, w_v, w_cp,
                                    w_res, ss, bs, ms, vs,
                                    wredh, beta_r, wqh, wkh, wvh, wcph, wresh, beta_s);
    // K1: transpose+cast x
    k_transpose<<<dim3(128, 8, 16), dim3(32, 8), 0, stream>>>(x, xt);
    // K2: reduce conv + BN + SiLU
    k_reduce<<<dim3(32, 16), 256, 0, stream>>>(xt, wredh, beta_r, xrt);
    // K3: 4x4 pool
    k_pool<<<1024, 256, 0, stream>>>(xrt, pooled);
    // K4: q (scaled by 1/sqrt(64)), feat in both layouts, k, v(channel-major)
    const long XRB = 4096 * 64, PB = 256 * 64;
    k_gemm64<<<dim3(32, 16), 256, 0, stream>>>(xrt, XRB, wqh, 0,
                                               qt, XRB, 64, 1, 4096, 0.125f);
    k_gemm64<<<dim3(32, 16), 256, 0, stream>>>(xrt, XRB, wcph, 0,
                                               cpt, XRB, 64, 1, 4096, 1.0f);
    k_gemm64<<<dim3(32, 16), 256, 0, stream>>>(xrt, XRB, wcph, 0,
                                               cpcm, XRB, 1, 4096, 4096, 1.0f);
    k_gemm64<<<dim3(2, 16), 256, 0, stream>>>(pooled, PB, wkh, 0,
                                              ktile, PB, 64, 1, 256, 1.0f);
    k_gemm64<<<dim3(2, 16), 256, 0, stream>>>(pooled, PB, wvh, 0,
                                              vcm, PB, 1, 256, 256, 1.0f);
    // K5: spatial attention -> fused[:, :64]
    k_spatial<<<dim3(32, 16), 256, 0, stream>>>(qt, ktile, vcm, fused);
    // K6/K7: channel affinity + softmax
    k_aff<<<32, 256, 0, stream>>>(cpcm, aff);
    k_affsm<<<4, 256, 0, stream>>>(aff, afth);
    // K8(channel): channel_t[n][c] = cpt @ afth^T -> fused[:, 64:]
    k_gemm64<<<dim3(32, 16), 256, 0, stream>>>(cpt, XRB, afth, (long)64 * 64,
                                               fused + 64, (long)4096 * 128, 128, 1,
                                               4096, 1.0f);
    // K9: restore conv + BN + residual
    k_restore<<<dim3(64, 16), 256, 0, stream>>>(fused, wresh, beta_s, x, gamma, out);
}
